// RoutingNet_56873956934312
// MI455X (gfx1250) — matrix operand order repaired and checked by reading, not compiled, not run on hardware
//
#include <hip/hip_runtime.h>
#include <cmath>

// ---------------------------------------------------------------------------
// RoutingNet for MI455X (gfx1250, wave32, WMMA).
// GEMM-shaped work runs on v_wmma_f32_16x16x32_f16 fed by vectorized,
// branch-free 16B/32B loads from pre-staged row-major f16 buffers (K padded
// to 32, rows padded/zeroed -> no predication in inner loops).
// Conv GEMM is blocked 16Mx64N per wave (one A fragment feeds 4 WMMAs).
// Fragment layouts used here (contiguous K order, not the ISA table order):
//   A 16x32 f16 : lane%16 = row M; half i -> K = grp*16 + i  (grp = lane/16)
//   B 32x16 f16 : lane%16 = col N; half i -> K = grp*16 + i
//   C/D f32     : VGPR r -> M = grp*8 + r, N = lane%16
// ---------------------------------------------------------------------------

typedef __attribute__((ext_vector_type(16))) _Float16 v16h;
typedef __attribute__((ext_vector_type(8)))  _Float16 h8;
typedef __attribute__((ext_vector_type(8)))  float    v8f;
typedef __attribute__((ext_vector_type(4)))  int      i32x4;

#define BN_RS 0.99999500003749978f /* rsqrt(1 + 1e-5) */

#if __has_builtin(__builtin_amdgcn_global_load_async_to_lds_b128) && \
    __has_builtin(__builtin_amdgcn_s_wait_asynccnt)
#define HAVE_ASYNC_LDS 1
#else
#define HAVE_ASYNC_LDS 0
#endif

__device__ __forceinline__ v16h make_a_frag(const _Float16* __restrict__ pa,
                                            int k0) {
  const h8 lo = *(const h8*)(pa + k0);
  const h8 hi = *(const h8*)(pa + k0 + 8);
  v16h af;
#pragma unroll
  for (int i = 0; i < 8; ++i) { af[i] = lo[i]; af[8 + i] = hi[i]; }
  return af;
}

// Single-tile (16x16) GEMM over row-major f16 operands, shared leading dim.
__device__ __forceinline__ v8f wmma_gemm(const _Float16* __restrict__ A, int rowA,
                                         const _Float16* __restrict__ B, int rowB,
                                         int K, int ldk) {
  const int lane = threadIdx.x & 31;
  const int grp  = lane >> 4;
  const _Float16* pa = A + (size_t)rowA * ldk + (grp << 4);
  const _Float16* pb = B + (size_t)rowB * ldk + (grp << 4);
  v8f acc = {};
  for (int k0 = 0; k0 < K; k0 += 32) {
    const v16h af = make_a_frag(pa, k0);
    const v16h bf = *(const v16h*)(pb + k0);
    if (k0 + 32 < K) {  // gfx1250 global_prefetch_b8
      __builtin_prefetch(pa + k0 + 32, 0, 0);
      __builtin_prefetch(pb + k0 + 32, 0, 0);
    }
    acc = __builtin_amdgcn_wmma_f32_16x16x32_f16(
        false, af, false, bf, (short)0, acc, false, false);
  }
  return acc;
}

// ---------------------------------------------------------------------------
// Staging / conversion kernels
// ---------------------------------------------------------------------------
__global__ __launch_bounds__(256) void cvt_rows_f16(
    const float* __restrict__ src, _Float16* __restrict__ dst, int rows_src,
    int cols, int rows_dst) {
  const int idx = blockIdx.x * blockDim.x + threadIdx.x;
  if (idx >= rows_dst * cols) return;
  const int r = idx / cols;
  dst[idx] = (_Float16)((r < rows_src) ? src[idx] : 0.f);
}

__global__ __launch_bounds__(256) void cvt_wgt_f16(
    const float* __restrict__ w, _Float16* __restrict__ W, int Co, int K,
    int K32) {
  const int idx = blockIdx.x * blockDim.x + threadIdx.x;
  if (idx >= Co * K32) return;
  const int co = idx / K32, k = idx - co * K32;
  W[idx] = (_Float16)((k < K) ? w[co * K + k] : 0.f);
}

// bilT[h][n][k] = bil_w[h][k][n]  (transpose so GEMM B loads are K-contiguous)
__global__ __launch_bounds__(256) void cvt_bilT_f16(
    const float* __restrict__ bw, _Float16* __restrict__ bt) {
  const int idx = blockIdx.x * blockDim.x + threadIdx.x;
  if (idx >= 100 * 256 * 256) return;
  const int k = idx & 255;
  int t = idx >> 8;
  const int n = t & 255;
  const int h = t >> 8;
  bt[idx] = (_Float16)bw[((size_t)h * 256 + k) * 256 + n];
}

__global__ __launch_bounds__(256) void zero_f16(_Float16* __restrict__ p, int n) {
  const int i = blockIdx.x * blockDim.x + threadIdx.x;
  if (i < n) p[i] = (_Float16)0.f;
}

// im2col patch matrix P[n][k] (f16), zero-padded to N64 x K32.
__global__ __launch_bounds__(256) void im2col_f16(
    const float* __restrict__ in, _Float16* __restrict__ P, int B, int Ci,
    int Hi, int Wi, int Ho, int Wo, int stride, int K, int K32, int N,
    int Npad) {
  const int idx = blockIdx.x * blockDim.x + threadIdx.x;
  if (idx >= Npad * K32) return;
  const int n = idx / K32, k = idx - n * K32;
  float v = 0.f;
  if (n < N && k < K) {
    const int ci = k / 9, r = k % 9, ky = r / 3, kx = r % 3;
    const int HoWo = Ho * Wo;
    const int b = n / HoWo, rr = n % HoWo, oy = rr / Wo, ox = rr % Wo;
    const int iy = oy * stride + ky - 1, ix = ox * stride + kx - 1;
    if (iy >= 0 && iy < Hi && ix >= 0 && ix < Wi)
      v = in[(((size_t)b * Ci + ci) * Hi + iy) * Wi + ix];
  }
  P[idx] = (_Float16)v;
}

// ---------------------------------------------------------------------------
// Block 0: conv(1->16,3x3,s2,p1)+BN+ReLU+pool2 fused (K=9: not a WMMA shape).
// ---------------------------------------------------------------------------
__global__ __launch_bounds__(256) void conv0_fused(
    const float* __restrict__ sup, const float* __restrict__ qry,
    const float* __restrict__ w, const float* __restrict__ g,
    const float* __restrict__ b, float* __restrict__ out) {
  int idx = blockIdx.x * blockDim.x + threadIdx.x;
  if (idx >= 250 * 16 * 32 * 32) return;
  const int ox = idx & 31;
  int t = idx >> 5;
  const int oy = t & 31; t >>= 5;
  const int co = t & 15; t >>= 4;
  const int img = t;
  const float* in = (img < 100) ? sup + (size_t)img * 16384
                                : qry + (size_t)(img - 100) * 16384;
  const float scale = g[co] * BN_RS;
  const float beta = b[co];
  const float* wk = w + co * 9;
  float mx = 0.0f;  // post-ReLU >= 0
#pragma unroll
  for (int py = 0; py < 2; ++py)
#pragma unroll
    for (int px = 0; px < 2; ++px) {
      const int cy = oy * 2 + py, cx = ox * 2 + px;
      float acc = 0.f;
#pragma unroll
      for (int ky = 0; ky < 3; ++ky) {
        const int iy = cy * 2 + ky - 1;
        if (iy < 0 || iy >= 128) continue;
#pragma unroll
        for (int kx = 0; kx < 3; ++kx) {
          const int ix = cx * 2 + kx - 1;
          if (ix < 0 || ix >= 128) continue;
          acc += wk[ky * 3 + kx] * in[iy * 128 + ix];
        }
      }
      mx = fmaxf(mx, fmaxf(acc * scale + beta, 0.f));
    }
  out[idx] = mx;
}

// ---------------------------------------------------------------------------
// Conv GEMM (blocks 1-4): A = wgtH (Co x K32), B = patchH (N64 x K32).
// 16Mx64N per wave: one A fragment feeds 4 WMMAs per K-step.
// blockIdx.y = M-tile (no integer division). Fused BN+ReLU store (NCHW).
// ---------------------------------------------------------------------------
__global__ __launch_bounds__(256) void gemm_conv_wmma(
    const _Float16* __restrict__ Ah, const _Float16* __restrict__ Bh,
    const float* __restrict__ gam, const float* __restrict__ bet,
    float* __restrict__ out, int Co, int K32, int N, int Ho, int Wo,
    int ntN4) {
  const int tM = blockIdx.y;
  const int tN4 = blockIdx.x * (blockDim.x >> 5) + (threadIdx.x >> 5);
  if (tN4 >= ntN4) return;  // wave-uniform; EXEC full at WMMA
  const int lane = threadIdx.x & 31, lm = lane & 15, grp = lane >> 4;
  const _Float16* pa = Ah + (size_t)(tM * 16 + lm) * K32 + (grp << 4);
  const _Float16* pb = Bh + (size_t)(tN4 * 64 + lm) * K32 + (grp << 4);
  const size_t rs16 = (size_t)16 * K32;
  v8f a0 = {}, a1 = {}, a2 = {}, a3 = {};
  for (int k0 = 0; k0 < K32; k0 += 32) {
    const v16h af = make_a_frag(pa, k0);
    const v16h b0 = *(const v16h*)(pb + k0);
    const v16h b1 = *(const v16h*)(pb + rs16 + k0);
    const v16h b2 = *(const v16h*)(pb + 2 * rs16 + k0);
    const v16h b3 = *(const v16h*)(pb + 3 * rs16 + k0);
    if (k0 + 32 < K32) {
      __builtin_prefetch(pa + k0 + 32, 0, 0);
      __builtin_prefetch(pb + k0 + 32, 0, 0);
      __builtin_prefetch(pb + rs16 + k0 + 32, 0, 0);
      __builtin_prefetch(pb + 2 * rs16 + k0 + 32, 0, 0);
      __builtin_prefetch(pb + 3 * rs16 + k0 + 32, 0, 0);
    }
    a0 = __builtin_amdgcn_wmma_f32_16x16x32_f16(false, af, false, b0, (short)0, a0, false, false);
    a1 = __builtin_amdgcn_wmma_f32_16x16x32_f16(false, af, false, b1, (short)0, a1, false, false);
    a2 = __builtin_amdgcn_wmma_f32_16x16x32_f16(false, af, false, b2, (short)0, a2, false, false);
    a3 = __builtin_amdgcn_wmma_f32_16x16x32_f16(false, af, false, b3, (short)0, a3, false, false);
  }
  const int HoWo = Ho * Wo;
#pragma unroll
  for (int s = 0; s < 4; ++s) {
    const v8f acc = (s == 0) ? a0 : (s == 1) ? a1 : (s == 2) ? a2 : a3;
    const int nIdx = tN4 * 64 + s * 16 + lm;
    if (nIdx < N) {
      const int b = nIdx / HoWo, rr = nIdx % HoWo, oy = rr / Wo, ox = rr % Wo;
#pragma unroll
      for (int r = 0; r < 8; ++r) {
        const int co = tM * 16 + (grp << 3) + r;
        const float x = acc[r] * (gam[co] * BN_RS) + bet[co];
        out[(((size_t)b * Co + co) * Ho + oy) * Wo + ox] = fmaxf(x, 0.f);
      }
    }
  }
}

__global__ __launch_bounds__(256) void maxpool2(const float* __restrict__ in,
                                                float* __restrict__ out,
                                                int BC, int H, int W) {
  const int Ho = H >> 1, Wo = W >> 1;
  const int idx = blockIdx.x * blockDim.x + threadIdx.x;
  if (idx >= BC * Ho * Wo) return;
  const int bc = idx / (Ho * Wo);
  const int r = idx % (Ho * Wo);
  const int oy = r / Wo, ox = r % Wo;
  const float* p = in + ((size_t)bc * H + oy * 2) * W + ox * 2;
  out[idx] = fmaxf(fmaxf(p[0], p[1]), fmaxf(p[W], p[W + 1]));
}

// ---------------------------------------------------------------------------
// e = sup(100x256) @ trans_w^T + trans_b   (blockIdx.y = M-tile)
// ---------------------------------------------------------------------------
__global__ __launch_bounds__(256) void gemm_e_wmma(
    const _Float16* __restrict__ fH, const _Float16* __restrict__ twH,
    const float* __restrict__ tb, float* __restrict__ e) {
  const int tM = blockIdx.y;  // 0..6
  const int tN = blockIdx.x * (blockDim.x >> 5) + (threadIdx.x >> 5);
  if (tN >= 16) return;
  const int lane = threadIdx.x & 31, lm = lane & 15, grp = lane >> 4;
  const int nB = tN * 16 + lm;
  const v8f acc = wmma_gemm(fH, tM * 16 + lm, twH, nB, 256, 256);
#pragma unroll
  for (int r = 0; r < 8; ++r) {
    const int m = tM * 16 + (grp << 3) + r;
    if (m < 100) e[m * 256 + nB] = acc[r] + tb[nB];
  }
}

// ---------------------------------------------------------------------------
// Dynamic routing (3 iters) over e(10,10,256) -> proto(10,256). One block.
// e is staged into LDS once via gfx1250 async global->LDS copies (ASYNCcnt),
// removing ~6 strided global re-reads of the 100KB tensor.
// ---------------------------------------------------------------------------
__global__ __launch_bounds__(256) void routing_kernel(
    const float* __restrict__ e, float* __restrict__ proto) {
  __shared__ __align__(16) float eL[25600];  // (10,10,256) f32 = 100KB LDS
  __shared__ float bb[100];
  __shared__ float coef[100];
  __shared__ float c[10 * 256];
  __shared__ float fac[10];
  const int t = threadIdx.x;
#if HAVE_ASYNC_LDS
  {
    typedef __attribute__((address_space(1))) i32x4* gp_t;
    typedef __attribute__((address_space(3))) i32x4* lp_t;
#pragma unroll
    for (int i = 0; i < 25; ++i) {  // 6400 16B chunks / 256 threads
      const int c4 = t + i * 256;
      __builtin_amdgcn_global_load_async_to_lds_b128(
          (gp_t)(e + c4 * 4), (lp_t)(eL + c4 * 4), 0, 0);
    }
    __builtin_amdgcn_s_wait_asynccnt(0);
  }
#else
  for (int i = t; i < 6400; i += 256)
    ((float4*)eL)[i] = ((const float4*)e)[i];
#endif
  if (t < 100) bb[t] = 0.f;
  __syncthreads();
  for (int iter = 0; iter < 3; ++iter) {
    if (t < 10) {
      float mx = bb[t * 10];
      for (int k = 1; k < 10; ++k) mx = fmaxf(mx, bb[t * 10 + k]);
      float tmp[10], se = 0.f;
      for (int k = 0; k < 10; ++k) { tmp[k] = expf(bb[t * 10 + k] - mx); se += tmp[k]; }
      for (int k = 0; k < 10; ++k) coef[t * 10 + k] = tmp[k] / se;
    }
    __syncthreads();
    for (int n = 0; n < 10; ++n) {
      float acc = 0.f;
      for (int k = 0; k < 10; ++k) acc += coef[n * 10 + k] * eL[(n * 10 + k) * 256 + t];
      c[n * 256 + t] = acc;
    }
    __syncthreads();
    if (t < 10) {
      float s2 = 0.f;
      for (int d = 0; d < 256; ++d) { const float x = c[t * 256 + d]; s2 += x * x; }
      fac[t] = sqrtf(s2) / (s2 + 1.0f);
    }
    __syncthreads();
    for (int n = 0; n < 10; ++n) c[n * 256 + t] *= fac[n];
    __syncthreads();
    if (iter < 2) {
      if (t < 100) {
        const int n = t / 10, k = t % 10;
        float acc = 0.f;
        for (int d = 0; d < 256; ++d) acc += c[n * 256 + d] * eL[(n * 10 + k) * 256 + d];
        bb[t] += acc;
      }
      __syncthreads();
    }
  }
  for (int n = 0; n < 10; ++n) proto[n * 256 + t] = c[n * 256 + t];
}

// ---------------------------------------------------------------------------
// t1[(n*100+h)][e] = proto[n,:] @ bilT[h][e][:]   (blockIdx.y = h; f16 out)
// ---------------------------------------------------------------------------
__global__ __launch_bounds__(256) void gemm_t1_wmma(
    const _Float16* __restrict__ protoH, const _Float16* __restrict__ bilT,
    _Float16* __restrict__ t1H) {
  const int h = blockIdx.y;  // 0..99
  const int tN = blockIdx.x * (blockDim.x >> 5) + (threadIdx.x >> 5);
  if (tN >= 16) return;
  const int lane = threadIdx.x & 31, lm = lane & 15, grp = lane >> 4;
  const int nB = tN * 16 + lm;
  const _Float16* bw = bilT + (size_t)h * 256 * 256;
  const v8f acc = wmma_gemm(protoH, lm, bw, nB, 256, 256);
#pragma unroll
  for (int r = 0; r < 8; ++r) {
    const int ml = (grp << 3) + r;
    if (ml < 10) t1H[((size_t)(ml * 100 + h)) * 256 + nB] = (_Float16)acc[r];
  }
}

// ---------------------------------------------------------------------------
// v[q, nh] = t1[nh,:] . qfeat[q,:]   (blockIdx.y = M-tile over nh)
// ---------------------------------------------------------------------------
__global__ __launch_bounds__(256) void gemm_v_wmma(
    const _Float16* __restrict__ t1H, const _Float16* __restrict__ fH,
    float* __restrict__ vout) {
  const int tM = blockIdx.y;  // 0..62
  const int tN = blockIdx.x * (blockDim.x >> 5) + (threadIdx.x >> 5);
  if (tN >= 10) return;
  const int lane = threadIdx.x & 31, lm = lane & 15, grp = lane >> 4;
  const int nQ = tN * 16 + lm;  // featsH rows 100+nQ <= 259 < 272 (zero pad)
  const v8f acc = wmma_gemm(t1H, tM * 16 + lm, fH + 100 * 256, nQ, 256, 256);
  if (nQ < 150) {
#pragma unroll
    for (int r = 0; r < 8; ++r) {
      const int m = tM * 16 + (grp << 3) + r;
      if (m < 1000) vout[(size_t)nQ * 1000 + m] = acc[r];
    }
  }
}

// ---------------------------------------------------------------------------
// s[q,n] = sum_h relu(v[q,n,h])*score_w[h] + score_b; out = log_softmax_n(s)
// ---------------------------------------------------------------------------
__global__ __launch_bounds__(256) void score_lsm(
    const float* __restrict__ v, const float* __restrict__ sw,
    const float* __restrict__ sb, float* __restrict__ out, int Q) {
  const int q = (blockIdx.x * blockDim.x + threadIdx.x) >> 5;
  const int lane = threadIdx.x & 31;
  if (q >= Q) return;
  float s[10];
#pragma unroll
  for (int n = 0; n < 10; ++n) {
    float acc = 0.f;
    for (int h = lane; h < 100; h += 32)
      acc += fmaxf(v[(size_t)q * 1000 + n * 100 + h], 0.f) * sw[h];
    for (int off = 16; off > 0; off >>= 1) acc += __shfl_xor(acc, off, 32);
    s[n] = acc + sb[0];
  }
  float mx = s[0];
#pragma unroll
  for (int n = 1; n < 10; ++n) mx = fmaxf(mx, s[n]);
  float se = 0.f;
#pragma unroll
  for (int n = 0; n < 10; ++n) se += expf(s[n] - mx);
  const float lz = mx + logf(se);
  if (lane < 10) out[q * 10 + lane] = s[lane] - lz;
}

// ---------------------------------------------------------------------------
extern "C" void kernel_launch(void* const* d_in, const int* in_sizes, int n_in,
                              void* d_out, int out_size, void* d_ws,
                              size_t ws_size, hipStream_t stream) {
  (void)in_sizes; (void)n_in; (void)out_size; (void)ws_size;
  const float* support = (const float*)d_in[0];
  const float* query   = (const float*)d_in[1];
  const float* cw[5] = {(const float*)d_in[2],  (const float*)d_in[5],
                        (const float*)d_in[8],  (const float*)d_in[11],
                        (const float*)d_in[14]};
  const float* bg[5] = {(const float*)d_in[3],  (const float*)d_in[6],
                        (const float*)d_in[9],  (const float*)d_in[12],
                        (const float*)d_in[15]};
  const float* bv[5] = {(const float*)d_in[4],  (const float*)d_in[7],
                        (const float*)d_in[10], (const float*)d_in[13],
                        (const float*)d_in[16]};
  const float* trans_w = (const float*)d_in[17];
  const float* trans_b = (const float*)d_in[18];
  const float* bil_w   = (const float*)d_in[19];
  const float* score_w = (const float*)d_in[20];
  const float* score_b = (const float*)d_in[21];

  auto blks = [](long n) { return (int)((n + 255) / 256); };

  // ---- workspace: f32 region then 32B-aligned f16 region -----------------
  float* p = (float*)d_ws;
  float* pool0 = p; p += 250 * 16 * 32 * 32;
  float* conv1 = p; p += 250 * 32 * 16 * 16;
  float* pool1 = p; p += 250 * 32 * 8 * 8;
  float* conv2 = p; p += 250 * 64 * 8 * 8;
  float* pool2 = p; p += 250 * 64 * 4 * 4;
  float* conv3 = p; p += 250 * 128 * 4 * 4;
  float* pool3 = p; p += 250 * 128 * 2 * 2;
  float* conv4 = p; p += 250 * 256 * 2 * 2;
  float* feats = p; p += 250 * 256;  // (250,256)
  float* ebuf  = p; p += 100 * 256;
  float* proto = p; p += 10 * 256;
  float* vbuf  = p; p += 150 * 1000 + 8;  // keep f16 region 32B aligned

  _Float16* h = (_Float16*)p;
  _Float16* patchH = h; h += 10240000;        // max N64*K32 (block 1)
  _Float16* wgtH   = h; h += 294912;          // max Co*K32 (block 4)
  _Float16* featsH = h; h += 272 * 256;       // rows 250..271 zero
  _Float16* twH    = h; h += 256 * 256;
  _Float16* protoH = h; h += 16 * 256;        // rows 10..15 zero
  _Float16* bilTH  = h; h += 100 * 256 * 256; // [h][n][k]
  _Float16* t1H    = h; h += 1008 * 256;      // rows 1000..1007 zero

  // ---- independent staging -----------------------------------------------
  cvt_rows_f16<<<blks(256 * 256), 256, 0, stream>>>(trans_w, twH, 256, 256, 256);
  cvt_bilT_f16<<<blks(100 * 256 * 256), 256, 0, stream>>>(bil_w, bilTH);
  zero_f16<<<blks(8 * 256), 256, 0, stream>>>(t1H + 1000 * 256, 8 * 256);

  // ---- block 0 (direct, fused pool) --------------------------------------
  conv0_fused<<<blks(250 * 16 * 32 * 32), 256, 0, stream>>>(
      support, query, cw[0], bg[0], bv[0], pool0);

  // ---- blocks 1-4: im2col -> WMMA GEMM (16x64 per wave) -> pool ----------
  struct Cfg { const float* in; float* cv; float* pl; int Ci, Hi, Co, st; };
  const Cfg cfg[4] = {
      {pool0, conv1, pool1, 16, 32, 32, 2},
      {pool1, conv2, pool2, 32, 8, 64, 1},
      {pool2, conv3, pool3, 64, 4, 128, 1},
      {pool3, conv4, feats, 128, 2, 256, 1},
  };
  for (int i = 0; i < 4; ++i) {
    const int B = 250, Ci = cfg[i].Ci, Hi = cfg[i].Hi, Wi = Hi;
    const int Co = cfg[i].Co, st = cfg[i].st;
    const int Ho = (st == 2) ? Hi / 2 : Hi, Wo = Ho;
    const int K = Ci * 9, K32 = (K + 31) / 32 * 32;
    const int N = B * Ho * Wo, N64 = (N + 63) / 64 * 64;
    cvt_wgt_f16<<<blks((long)Co * K32), 256, 0, stream>>>(cw[i + 1], wgtH, Co, K, K32);
    im2col_f16<<<blks((long)N64 * K32), 256, 0, stream>>>(
        cfg[i].in, patchH, B, Ci, Hi, Wi, Ho, Wo, st, K, K32, N, N64);
    const int ntN4 = N64 / 64;
    dim3 grid((ntN4 + 7) / 8, Co / 16);
    gemm_conv_wmma<<<grid, 256, 0, stream>>>(wgtH, patchH, bg[i + 1], bv[i + 1],
                                             cfg[i].cv, Co, K32, N, Ho, Wo, ntN4);
    maxpool2<<<blks((long)B * Co * (Ho / 2) * (Wo / 2)), 256, 0, stream>>>(
        cfg[i].cv, cfg[i].pl, B * Co, Ho, Wo);
  }

  // ---- head ---------------------------------------------------------------
  cvt_rows_f16<<<blks(272 * 256), 256, 0, stream>>>(feats, featsH, 250, 256, 272);
  gemm_e_wmma<<<dim3(2, 7), 256, 0, stream>>>(featsH, twH, trans_b, ebuf);
  routing_kernel<<<1, 256, 0, stream>>>(ebuf, proto);
  cvt_rows_f16<<<blks(16 * 256), 256, 0, stream>>>(proto, protoH, 10, 256, 16);
  gemm_t1_wmma<<<dim3(2, 100), 256, 0, stream>>>(protoH, bilTH, t1H);
  gemm_v_wmma<<<dim3(2, 63), 256, 0, stream>>>(t1H, featsH, vbuf);
  score_lsm<<<(150 * 32 + 255) / 256, 256, 0, stream>>>(vbuf, score_w, score_b,
                                                        (float*)d_out, 150);
}
